// TransformerBlock_49005576847634
// MI455X (gfx1250) — matrix operand order repaired and checked by reading, not compiled, not run on hardware
//
#include <hip/hip_runtime.h>
#include <math.h>

// ---------------------------------------------------------------------------
// Transformer block (B=4, T=2048, D=1024, H=16, dh=64, F=4096), fp32 I/O.
// Compute-bound (~343 GFLOP) -> all GEMMs + attention via
// v_wmma_f32_16x16x32_bf16 (fp32 accumulate); LN/softmax/residuals in fp32.
// GEMM tiles staged with global_load_async_to_lds_b128 (ASYNCcnt) into
// double-buffered LDS so WMMA overlaps the next tile fill.
// ---------------------------------------------------------------------------

typedef __attribute__((ext_vector_type(16))) __bf16 v16bf;
typedef __attribute__((ext_vector_type(8)))  float  v8f;
typedef __attribute__((ext_vector_type(4)))  unsigned int u32x4;
typedef __attribute__((ext_vector_type(4)))  float  f32x4;

#define TB_B 4
#define TB_T 2048
#define TB_D 1024
#define TB_H 16
#define TB_DH 64
#define TB_F 4096
#define TB_M (TB_B * TB_T)

union Frag {
    v16bf v;
    u32x4 q[2];
    unsigned short s[16];
};

__device__ __forceinline__ unsigned short f2bfu(float f) {
    unsigned u = __builtin_bit_cast(unsigned, f);
    u += 0x7FFFu + ((u >> 16) & 1u);      // round-to-nearest-even
    return (unsigned short)(u >> 16);
}

// ---------------------------------------------------------------------------
// fp32 -> bf16 weight conversion
// ---------------------------------------------------------------------------
__global__ __launch_bounds__(256) void cvt_bf16_kernel(
    const float* __restrict__ in, unsigned short* __restrict__ out, int n)
{
    const int i = (blockIdx.x * 256 + threadIdx.x) * 4;
    if (i >= n) return;
    f32x4 v = *(const f32x4*)(in + i);
    out[i + 0] = f2bfu(v.x);
    out[i + 1] = f2bfu(v.y);
    out[i + 2] = f2bfu(v.z);
    out[i + 3] = f2bfu(v.w);
}

// ---------------------------------------------------------------------------
// LayerNorm over D=1024, one row per 256-thread block, bf16 output
// ---------------------------------------------------------------------------
__global__ __launch_bounds__(256) void ln_kernel(
    const float* __restrict__ x, const float* __restrict__ g,
    const float* __restrict__ bta, unsigned short* __restrict__ out)
{
    const int row = blockIdx.x;
    const int tid = threadIdx.x;
    f32x4 v = *(const f32x4*)(x + (size_t)row * TB_D + tid * 4);
    float s  = v.x + v.y + v.z + v.w;
    float sq = v.x * v.x + v.y * v.y + v.z * v.z + v.w * v.w;
#pragma unroll
    for (int off = 16; off > 0; off >>= 1) {
        s  += __shfl_xor(s,  off);
        sq += __shfl_xor(sq, off);
    }
    __shared__ float rs[8], rq[8];
    const int wave = tid >> 5, lane = tid & 31;
    if (lane == 0) { rs[wave] = s; rq[wave] = sq; }
    __syncthreads();
    float ts = 0.f, tq = 0.f;
#pragma unroll
    for (int i = 0; i < 8; ++i) { ts += rs[i]; tq += rq[i]; }
    const float mu   = ts * (1.0f / TB_D);
    const float var  = tq * (1.0f / TB_D) - mu * mu;
    const float rstd = rsqrtf(var + 1e-5f);
    const int c0 = tid * 4;
    out[(size_t)row * TB_D + c0 + 0] = f2bfu((v.x - mu) * rstd * g[c0 + 0] + bta[c0 + 0]);
    out[(size_t)row * TB_D + c0 + 1] = f2bfu((v.y - mu) * rstd * g[c0 + 1] + bta[c0 + 1]);
    out[(size_t)row * TB_D + c0 + 2] = f2bfu((v.z - mu) * rstd * g[c0 + 2] + bta[c0 + 2]);
    out[(size_t)row * TB_D + c0 + 3] = f2bfu((v.w - mu) * rstd * g[c0 + 3] + bta[c0 + 3]);
}

// ---------------------------------------------------------------------------
// bf16 GEMM: C[M,N] = A[M,K] @ W[K,N] (+bias, epilogue variants)
//   EPI 0: bf16 out, (acc+bias)*scale        (Q/K projections)
//   EPI 1: bf16 out, transposed per head     (V -> (B,H,dh,T))
//   EPI 2: bf16 out, exact GELU(acc+bias)    (FFN1)
//   EPI 3: f32 out,  acc+bias+resid          (out-proj / FFN2 residual)
// Block tile 128x128, BK=32 (= one WMMA K). 8 waves: 2(M) x 4(N), each 64x32.
// Double-buffered LDS filled with global_load_async_to_lds_b128; one
// s_wait_asynccnt + one barrier per K-step.
// ---------------------------------------------------------------------------
template <int EPI>
__global__ __launch_bounds__(256) void gemm_bf16_kernel(
    const unsigned short* __restrict__ A, const unsigned short* __restrict__ W,
    const float* __restrict__ bias, const float* __restrict__ resid,
    void* __restrict__ outp, int Mdim, int Ndim, int Kdim, float scale)
{
    __shared__ __align__(16) unsigned short As[2][128 * 40];   // [m][k], pad 8
    __shared__ __align__(16) unsigned short Bs[2][32 * 136];   // [k][n], pad 8

    const int tid  = threadIdx.x;
    const int wave = tid >> 5, lane = tid & 31;
    const int hi   = lane >> 4, l16 = lane & 15;
    const int wm0  = (wave >> 2) * 64;
    const int wn0  = (wave & 3) * 32;
    const int m0   = blockIdx.y * 128;
    const int n0   = blockIdx.x * 128;

    // Async-stage one 128x32 A tile + 32x128 B tile into LDS buffer `buf`.
    auto stage = [&](int buf, int k0) {
#pragma unroll
        for (int i = 0; i < 2; ++i) {          // A: 512 16B chunks, 2/thread
            const int c = tid * 2 + i;
            const int row = c >> 2, kc = (c & 3) * 8;
            const unsigned long long ga =
                (unsigned long long)(A + (size_t)(m0 + row) * Kdim + k0 + kc);
            const unsigned la = (unsigned)(size_t)&As[buf][row * 40 + kc];
            asm volatile("global_load_async_to_lds_b128 %0, %1, off"
                         :: "v"(la), "v"(ga) : "memory");
        }
#pragma unroll
        for (int i = 0; i < 2; ++i) {          // B: 512 16B chunks, 2/thread
            const int c = tid * 2 + i;
            const int kr = c >> 4, nc = (c & 15) * 8;
            const unsigned long long ga =
                (unsigned long long)(W + (size_t)(k0 + kr) * Ndim + n0 + nc);
            const unsigned la = (unsigned)(size_t)&Bs[buf][kr * 136 + nc];
            asm volatile("global_load_async_to_lds_b128 %0, %1, off"
                         :: "v"(la), "v"(ga) : "memory");
        }
    };

    const v8f vzero = {0.f, 0.f, 0.f, 0.f, 0.f, 0.f, 0.f, 0.f};
    v8f acc[4][2];
#pragma unroll
    for (int mt = 0; mt < 4; ++mt)
#pragma unroll
        for (int nt = 0; nt < 2; ++nt) acc[mt][nt] = vzero;

    const int nk = Kdim >> 5;
    stage(0, 0);
    for (int ks = 0; ks < nk; ++ks) {
        // Drain this wave's async fills, then rendezvous: current buffer full.
        asm volatile("s_wait_asynccnt 0" ::: "memory");
        __syncthreads();
        if (ks + 1 < nk) stage((ks + 1) & 1, (ks + 1) << 5);  // overlap next fill

        const int buf = ks & 1;
        // A frag: elem j -> K = (j/8)*16 + hi*8 + j%8  (two b128 LDS reads)
        Frag af[4];
#pragma unroll
        for (int mt = 0; mt < 4; ++mt) {
            const unsigned short* p = &As[buf][(wm0 + mt * 16 + l16) * 40 + hi * 8];
            af[mt].q[0] = *(const u32x4*)p;
            af[mt].q[1] = *(const u32x4*)(p + 16);
        }
        // B frag: elem j -> K = (j/8)*16 + hi*8 + j%8, col = l16 (gather from [k][n] tile)
        Frag bfr[2];
#pragma unroll
        for (int nt = 0; nt < 2; ++nt) {
            const int n = wn0 + nt * 16 + l16;
#pragma unroll
            for (int j = 0; j < 16; ++j)
                bfr[nt].s[j] = Bs[buf][((j >> 3) * 16 + hi * 8 + (j & 7)) * 136 + n];
        }
#pragma unroll
        for (int mt = 0; mt < 4; ++mt)
#pragma unroll
            for (int nt = 0; nt < 2; ++nt)
                acc[mt][nt] = __builtin_amdgcn_wmma_f32_16x16x32_bf16(
                    false, af[mt].v, false, bfr[nt].v, (short)0, acc[mt][nt],
                    false, false);
        // No trailing barrier: frag ds_loads are consumed by WMMA (dscnt
        // waits) before any wave reaches the next top-of-loop barrier, and
        // the in-flight async fill targets the other buffer.
    }

    // Epilogue. C layout: VGPR r -> M = r + 8*hi, N = l16.
#pragma unroll
    for (int mt = 0; mt < 4; ++mt) {
#pragma unroll
        for (int nt = 0; nt < 2; ++nt) {
#pragma unroll
            for (int r = 0; r < 8; ++r) {
                const int row = m0 + wm0 + mt * 16 + r + 8 * hi;
                const int col = n0 + wn0 + nt * 16 + l16;
                const float v = acc[mt][nt][r] + bias[col];
                if (EPI == 0) {
                    ((unsigned short*)outp)[(size_t)row * Ndim + col] = f2bfu(v * scale);
                } else if (EPI == 1) {
                    const int bb = row >> 11;          // row / T
                    const int tt = row & (TB_T - 1);
                    const int hh = col >> 6;           // col / dh
                    const int dd = col & (TB_DH - 1);
                    ((unsigned short*)outp)[(((size_t)(bb * TB_H + hh)) * TB_DH + dd) * TB_T + tt] =
                        f2bfu(v);
                } else if (EPI == 2) {
                    const float gl = 0.5f * v * (1.0f + erff(v * 0.70710678118654752f));
                    ((unsigned short*)outp)[(size_t)row * Ndim + col] = f2bfu(gl);
                } else {
                    ((float*)outp)[(size_t)row * Ndim + col] =
                        v + resid[(size_t)row * Ndim + col];
                }
            }
        }
    }
}

// ---------------------------------------------------------------------------
// Causal flash attention. One wave handles 16 query rows of one (b,h).
// q,k: (B*T, D) bf16 (q pre-scaled by 1/8); vt: (B,H,dh,T) bf16;
// out: (B*T, D) bf16. Block = 4 waves (independent query tiles).
// ---------------------------------------------------------------------------
__global__ __launch_bounds__(128) void attn_kernel(
    const unsigned short* __restrict__ qm, const unsigned short* __restrict__ km,
    const unsigned short* __restrict__ vt, unsigned short* __restrict__ om)
{
    __shared__ __align__(16) unsigned short Ps[4][16][40];  // per-wave P tile

    const int tid  = threadIdx.x;
    const int wave = tid >> 5, lane = tid & 31;
    const int hi   = lane >> 4, l16 = lane & 15;
    const int qt   = blockIdx.x * 4 + wave;
    const int q0   = qt * 16;
    const int bh   = blockIdx.y;
    const int b    = bh >> 4, hh = bh & (TB_H - 1);

    // Q fragments: two K=32 chunks over dh=64.
    Frag Qf[2];
    {
        const unsigned short* qbase = qm + (size_t)(b * TB_T + q0 + l16) * TB_D + hh * TB_DH;
#pragma unroll
        for (int c = 0; c < 2; ++c) {
            const unsigned short* p = qbase + c * 32 + hi * 8;
            Qf[c].q[0] = *(const u32x4*)p;
            Qf[c].q[1] = *(const u32x4*)(p + 16);
        }
    }

    const v8f vzero = {0.f, 0.f, 0.f, 0.f, 0.f, 0.f, 0.f, 0.f};
    v8f O[4];
    float mr[8], lr[8];
#pragma unroll
    for (int t = 0; t < 4; ++t) O[t] = vzero;
#pragma unroll
    for (int r = 0; r < 8; ++r) { mr[r] = -1e30f; lr[r] = 0.f; }

    for (int kb = 0; kb <= q0; kb += 32) {
        const bool do1 = (kb + 16 <= q0);
        v8f S0 = vzero, S1 = vzero;
#pragma unroll
        for (int c = 0; c < 2; ++c) {
            Frag Kf;
            const unsigned short* kp =
                km + (size_t)(b * TB_T + kb + l16) * TB_D + hh * TB_DH + c * 32 + hi * 8;
            Kf.q[0] = *(const u32x4*)kp;
            Kf.q[1] = *(const u32x4*)(kp + 16);
            S0 = __builtin_amdgcn_wmma_f32_16x16x32_bf16(
                false, Qf[c].v, false, Kf.v, (short)0, S0, false, false);
            if (do1) {
                Frag Kg;
                const unsigned short* kp2 = kp + (size_t)16 * TB_D;
                Kg.q[0] = *(const u32x4*)kp2;
                Kg.q[1] = *(const u32x4*)(kp2 + 16);
                S1 = __builtin_amdgcn_wmma_f32_16x16x32_bf16(
                    false, Qf[c].v, false, Kg.v, (short)0, S1, false, false);
            }
        }
        // Online softmax update (fp32). Row M = r + 8*hi lives in one 16-lane
        // half-group, so xor-shuffles 1..8 reduce exactly over its 16 columns.
        float sc[8];
#pragma unroll
        for (int r = 0; r < 8; ++r) {
            const int qidx = q0 + r + 8 * hi;
            if (kb + l16 > qidx) S0[r] = -1e30f;
            if (do1 && (kb + 16 + l16 > qidx)) S1[r] = -1e30f;
            float s = do1 ? fmaxf(S0[r], S1[r]) : S0[r];
#pragma unroll
            for (int off = 1; off < 16; off <<= 1) s = fmaxf(s, __shfl_xor(s, off));
            const float mnew = fmaxf(mr[r], s);
            const float resc = __expf(mr[r] - mnew);
            const float e0 = __expf(S0[r] - mnew);
            float rowsum = e0;
            Ps[wave][r + 8 * hi][l16] = f2bfu(e0);
            float e1 = 0.f;
            if (do1) e1 = __expf(S1[r] - mnew);
            Ps[wave][r + 8 * hi][16 + l16] = f2bfu(e1);
            rowsum += e1;
#pragma unroll
            for (int off = 1; off < 16; off <<= 1) rowsum += __shfl_xor(rowsum, off);
            lr[r] = lr[r] * resc + rowsum;
            mr[r] = mnew;
            sc[r] = resc;
        }
#pragma unroll
        for (int t = 0; t < 4; ++t)
#pragma unroll
            for (int r = 0; r < 8; ++r) O[t][r] *= sc[r];

        // Same-wave LDS store->load (cross-lane): drain DS before reading.
        asm volatile("s_wait_dscnt 0" ::: "memory");

        Frag Pf;
        const unsigned short* pp = &Ps[wave][l16][hi * 8];
        Pf.q[0] = *(const u32x4*)pp;
        Pf.q[1] = *(const u32x4*)(pp + 16);
#pragma unroll
        for (int nt = 0; nt < 4; ++nt) {
            Frag Vf;  // B frag: contraction = key kb + (j/8)*16 + hi*8 + j%8, col = dh
            const unsigned short* vp =
                vt + ((size_t)(b * TB_H + hh) * TB_DH + nt * 16 + l16) * TB_T + kb + hi * 8;
            Vf.q[0] = *(const u32x4*)vp;
            Vf.q[1] = *(const u32x4*)(vp + 16);
            O[nt] = __builtin_amdgcn_wmma_f32_16x16x32_bf16(
                false, Pf.v, false, Vf.v, (short)0, O[nt], false, false);
        }
    }

    // Normalize and write back as (B,T,H*dh) bf16.
#pragma unroll
    for (int nt = 0; nt < 4; ++nt)
#pragma unroll
        for (int r = 0; r < 8; ++r) {
            const int t = q0 + r + 8 * hi;
            const int d = nt * 16 + l16;
            om[(size_t)(b * TB_T + t) * TB_D + hh * TB_DH + d] = f2bfu(O[nt][r] / lr[r]);
        }
}

// ---------------------------------------------------------------------------
// Host-side orchestration
// ---------------------------------------------------------------------------
extern "C" void kernel_launch(void* const* d_in, const int* in_sizes, int n_in,
                              void* d_out, int out_size, void* d_ws, size_t ws_size,
                              hipStream_t stream)
{
    (void)in_sizes; (void)n_in; (void)out_size; (void)ws_size;
    const float* x    = (const float*)d_in[0];
    const float* ln1g = (const float*)d_in[1];
    const float* ln1b = (const float*)d_in[2];
    const float* wq   = (const float*)d_in[3];
    const float* bq   = (const float*)d_in[4];
    const float* wk   = (const float*)d_in[5];
    const float* bk   = (const float*)d_in[6];
    const float* wv   = (const float*)d_in[7];
    const float* bv   = (const float*)d_in[8];
    const float* wo   = (const float*)d_in[9];
    const float* bo   = (const float*)d_in[10];
    const float* ln2g = (const float*)d_in[11];
    const float* ln2b = (const float*)d_in[12];
    const float* w1   = (const float*)d_in[13];
    const float* b1   = (const float*)d_in[14];
    const float* w2   = (const float*)d_in[15];
    const float* b2   = (const float*)d_in[16];

    char* ws = (char*)d_ws;
    size_t off = 0;
    auto alloc = [&](size_t bytes) -> void* {
        void* p = ws + off;
        off = (off + bytes + 255) & ~(size_t)255;
        return p;
    };
    unsigned short* h_bf   = (unsigned short*)alloc((size_t)TB_M * TB_D * 2);
    unsigned short* wq_bf  = (unsigned short*)alloc((size_t)TB_D * TB_D * 2);
    unsigned short* wk_bf  = (unsigned short*)alloc((size_t)TB_D * TB_D * 2);
    unsigned short* wv_bf  = (unsigned short*)alloc((size_t)TB_D * TB_D * 2);
    unsigned short* wo_bf  = (unsigned short*)alloc((size_t)TB_D * TB_D * 2);
    unsigned short* w1_bf  = (unsigned short*)alloc((size_t)TB_D * TB_F * 2);
    unsigned short* w2_bf  = (unsigned short*)alloc((size_t)TB_F * TB_D * 2);
    unsigned short* q_bf   = (unsigned short*)alloc((size_t)TB_M * TB_D * 2);
    unsigned short* k_bf   = (unsigned short*)alloc((size_t)TB_M * TB_D * 2);
    unsigned short* vt_bf  = (unsigned short*)alloc((size_t)TB_M * TB_D * 2);
    unsigned short* at_bf  = (unsigned short*)alloc((size_t)TB_M * TB_D * 2);
    float*          x1     = (float*)alloc((size_t)TB_M * TB_D * 4);
    unsigned short* h2_bf  = (unsigned short*)alloc((size_t)TB_M * TB_D * 2);
    unsigned short* a1_bf  = (unsigned short*)alloc((size_t)TB_M * TB_F * 2);

    // 1) weights -> bf16
    cvt_bf16_kernel<<<TB_D * TB_D / 1024, 256, 0, stream>>>(wq, wq_bf, TB_D * TB_D);
    cvt_bf16_kernel<<<TB_D * TB_D / 1024, 256, 0, stream>>>(wk, wk_bf, TB_D * TB_D);
    cvt_bf16_kernel<<<TB_D * TB_D / 1024, 256, 0, stream>>>(wv, wv_bf, TB_D * TB_D);
    cvt_bf16_kernel<<<TB_D * TB_D / 1024, 256, 0, stream>>>(wo, wo_bf, TB_D * TB_D);
    cvt_bf16_kernel<<<TB_D * TB_F / 1024, 256, 0, stream>>>(w1, w1_bf, TB_D * TB_F);
    cvt_bf16_kernel<<<TB_F * TB_D / 1024, 256, 0, stream>>>(w2, w2_bf, TB_F * TB_D);

    // 2) LN1
    ln_kernel<<<TB_M, 256, 0, stream>>>(x, ln1g, ln1b, h_bf);

    // 3) Q/K/V projections (q pre-scaled by 1/sqrt(dh); v head-transposed)
    dim3 gD(TB_D / 128, TB_M / 128);
    gemm_bf16_kernel<0><<<gD, 256, 0, stream>>>(h_bf, wq_bf, bq, nullptr, q_bf,
                                                TB_M, TB_D, TB_D, 0.125f);
    gemm_bf16_kernel<0><<<gD, 256, 0, stream>>>(h_bf, wk_bf, bk, nullptr, k_bf,
                                                TB_M, TB_D, TB_D, 1.0f);
    gemm_bf16_kernel<1><<<gD, 256, 0, stream>>>(h_bf, wv_bf, bv, nullptr, vt_bf,
                                                TB_M, TB_D, TB_D, 1.0f);

    // 4) causal flash attention
    attn_kernel<<<dim3(TB_T / 64, TB_B * TB_H), 128, 0, stream>>>(q_bf, k_bf, vt_bf, at_bf);

    // 5) out-proj + residual (fp32)
    gemm_bf16_kernel<3><<<gD, 256, 0, stream>>>(at_bf, wo_bf, bo, x, x1,
                                                TB_M, TB_D, TB_D, 1.0f);

    // 6) LN2
    ln_kernel<<<TB_M, 256, 0, stream>>>(x1, ln2g, ln2b, h2_bf);

    // 7) FFN1 + GELU, FFN2 + residual -> d_out
    dim3 gF(TB_F / 128, TB_M / 128);
    gemm_bf16_kernel<2><<<gF, 256, 0, stream>>>(h2_bf, w1_bf, b1, nullptr, a1_bf,
                                                TB_M, TB_F, TB_D, 1.0f);
    gemm_bf16_kernel<3><<<gD, 256, 0, stream>>>(a1_bf, w2_bf, b2, x1, d_out,
                                                TB_M, TB_D, TB_F, 1.0f);
}
